// MutualInformation_52243982188892
// MI455X (gfx1250) — matrix operand order repaired and checked by reading, not compiled, not run on hardware
//
#include <hip/hip_runtime.h>

typedef __attribute__((ext_vector_type(16))) _Float16 v16h;
typedef __attribute__((ext_vector_type(8)))  _Float16 v8h;
typedef __attribute__((ext_vector_type(2)))  _Float16 v2h;
typedef __attribute__((ext_vector_type(8)))  float    v8f;

#define BATCH 4
#define EPS_NORM 1e-8f
#define EPS_LOG  1e-10f

// zero-instruction compiler ordering fence: per-wave LDS ops are in-order in HW
#define LDS_ORDER_FENCE() asm volatile("" ::: "memory")

// ---- order-preserving float <-> uint keys (for atomic min/max) ----
__device__ __forceinline__ unsigned f2key(float f) {
  unsigned u = __float_as_uint(f);
  return (u & 0x80000000u) ? ~u : (u | 0x80000000u);
}
__device__ __forceinline__ float key2f(unsigned k) {
  unsigned u = (k & 0x80000000u) ? (k & 0x7FFFFFFFu) : ~k;
  return __uint_as_float(u);
}
__device__ __forceinline__ unsigned kmin(unsigned a, unsigned b) { return a < b ? a : b; }
__device__ __forceinline__ unsigned kmax(unsigned a, unsigned b) { return a > b ? a : b; }

// ---- kernel 0: reset workspace every launch (ws is not re-poisoned) ----
__global__ void mi_init(unsigned* mm, float* hists, int nfloats) {
  int t = threadIdx.x;
  if (t < BATCH * 4) mm[t] = (t & 1) ? 0u : 0xFFFFFFFFu;  // [min,max] key pairs
  for (int i = t; i < nfloats; i += blockDim.x) hists[i] = 0.0f;
}

// ---- kernel 1: per-batch min/max of fixed & warped ----
__global__ void __launch_bounds__(256) mi_minmax(const float* __restrict__ fixed,
                                                 const float* __restrict__ warped,
                                                 unsigned* __restrict__ mm, int n) {
  const int b = blockIdx.y;
  const float4* f4 = (const float4*)(fixed + (size_t)b * n);
  const float4* w4 = (const float4*)(warped + (size_t)b * n);
  const int n4 = n >> 2;
  unsigned fmn = 0xFFFFFFFFu, fmx = 0u, wmn = 0xFFFFFFFFu, wmx = 0u;
  for (int i = blockIdx.x * blockDim.x + threadIdx.x; i < n4; i += gridDim.x * blockDim.x) {
    float4 a = f4[i], c = w4[i];
    unsigned k0 = f2key(a.x), k1 = f2key(a.y), k2 = f2key(a.z), k3 = f2key(a.w);
    fmn = kmin(fmn, kmin(kmin(k0, k1), kmin(k2, k3)));
    fmx = kmax(fmx, kmax(kmax(k0, k1), kmax(k2, k3)));
    k0 = f2key(c.x); k1 = f2key(c.y); k2 = f2key(c.z); k3 = f2key(c.w);
    wmn = kmin(wmn, kmin(kmin(k0, k1), kmin(k2, k3)));
    wmx = kmax(wmx, kmax(kmax(k0, k1), kmax(k2, k3)));
  }
  __shared__ unsigned s0[256], s1[256], s2[256], s3[256];
  const int t = threadIdx.x;
  s0[t] = fmn; s1[t] = fmx; s2[t] = wmn; s3[t] = wmx;
  __syncthreads();
  for (int s = 128; s > 0; s >>= 1) {
    if (t < s) {
      s0[t] = kmin(s0[t], s0[t + s]);
      s1[t] = kmax(s1[t], s1[t + s]);
      s2[t] = kmin(s2[t], s2[t + s]);
      s3[t] = kmax(s3[t], s3[t + s]);
    }
    __syncthreads();
  }
  if (t == 0) {
    atomicMin(&mm[b * 4 + 0], s0[0]);
    atomicMax(&mm[b * 4 + 1], s1[0]);
    atomicMin(&mm[b * 4 + 2], s2[0]);
    atomicMax(&mm[b * 4 + 3], s3[0]);
  }
}

// ---- kernel 2: fused weight-gen + WMMA histogram accumulation ----
// Staging tile is BIN-major [16 bins][64 voxels] f16 so that:
//   producer: lane owns voxels (2l,2l+1) -> packed v2h b32 stores per bin row
//   consumer: A/B fragments are contiguous row segments -> ds_load_b128 x2 each
// Per 64-voxel iteration: 2 sub-chunks of K=32 -> 6x v_wmma_f32_16x16x32_f16
//   joint   : D += Wf^T x Ww
//   marg  f : D += Wf^T x ones   (hist_f replicated down columns)
//   marg  w : D += ones x Ww     (hist_w replicated across rows)
__global__ void __launch_bounds__(256) mi_hist(const float* __restrict__ fixed,
                                               const float* __restrict__ warped,
                                               const unsigned* __restrict__ mm,
                                               float* __restrict__ histj,
                                               float* __restrict__ histf,
                                               float* __restrict__ histw, int n) {
  const int b = blockIdx.y;
  const int lane = threadIdx.x & 31;
  const int wave = threadIdx.x >> 5;
  const int halfSel = lane >> 4;  // which 16-lane half of the wave
  const int m = lane & 15;        // row/col index within half

  // 16 bins x 64 voxels, padded to 72 halves (stride 36 dwords: 36m % 64 distinct)
  __shared__ __align__(16) _Float16 tf[8][16][72];
  __shared__ __align__(16) _Float16 tw[8][16][72];

  const float fmin = key2f(mm[b * 4 + 0]);
  const float fscale = 1.0f / (key2f(mm[b * 4 + 1]) - fmin + EPS_NORM);
  const float wmin = key2f(mm[b * 4 + 2]);
  const float wscale = 1.0f / (key2f(mm[b * 4 + 3]) - wmin + EPS_NORM);

  // Gaussian factorization: w_k = exp(-x^2/2) * exp(x/15)^k * exp(-(k/15)^2/2)
  float ck[16];
#pragma unroll
  for (int k = 0; k < 16; k++) {
    float bk = (float)k * (1.0f / 15.0f);
    ck[k] = __expf(-0.5f * bk * bk);
  }

  v16h ones;
#pragma unroll
  for (int i = 0; i < 16; i++) ones[i] = (_Float16)1.0f;

  v8f cj = {};  // joint 16x16 accumulator
  v8f cf = {};  // marginal fixed (replicated cols)
  v8f cw = {};  // marginal warped (replicated rows)

  const float2* fb2 = (const float2*)(fixed + (size_t)b * n);
  const float2* wb2 = (const float2*)(warped + (size_t)b * n);
  const int nChunks = n >> 6;  // 64 voxels per wave-iteration
  const int wavesTotal = gridDim.x * (blockDim.x >> 5);

  for (int c = blockIdx.x * (blockDim.x >> 5) + wave; c < nChunks; c += wavesTotal) {
    const int idx2 = (c << 5) + lane;  // float2 index; voxels 2*idx2, 2*idx2+1
    const float2 xf2 = fb2[idx2];
    const float2 xw2 = wb2[idx2];
    // speculative prefetch of next iteration's cachelines (global_prefetch_b8)
    __builtin_prefetch(&fb2[idx2 + (wavesTotal << 5)], 0, 0);
    __builtin_prefetch(&wb2[idx2 + (wavesTotal << 5)], 0, 0);

    const float xf0 = (xf2.x - fmin) * fscale, xf1 = (xf2.y - fmin) * fscale;
    const float xw0 = (xw2.x - wmin) * wscale, xw1 = (xw2.y - wmin) * wscale;
    float pf0 = __expf(-0.5f * xf0 * xf0), tf0 = __expf(xf0 * (1.0f / 15.0f));
    float pf1 = __expf(-0.5f * xf1 * xf1), tf1 = __expf(xf1 * (1.0f / 15.0f));
    float pw0 = __expf(-0.5f * xw0 * xw0), tw0 = __expf(xw0 * (1.0f / 15.0f));
    float pw1 = __expf(-0.5f * xw1 * xw1), tw1 = __expf(xw1 * (1.0f / 15.0f));

#pragma unroll
    for (int k = 0; k < 16; k++) {  // packed pair (voxel 2l, 2l+1) per bin row
      v2h pkf = { (_Float16)(pf0 * ck[k]), (_Float16)(pf1 * ck[k]) };
      v2h pkw = { (_Float16)(pw0 * ck[k]), (_Float16)(pw1 * ck[k]) };
      *(v2h*)&tf[wave][k][lane << 1] = pkf;
      *(v2h*)&tw[wave][k][lane << 1] = pkw;
      pf0 *= tf0; pf1 *= tf1; pw0 *= tw0; pw1 *= tw1;
    }
    LDS_ORDER_FENCE();  // HW keeps per-wave LDS in order; pin compiler ordering

    const v8h* rowF = (const v8h*)&tf[wave][m][0];  // 9 v8h per padded row
    const v8h* rowW = (const v8h*)&tw[wave][m][0];
#pragma unroll
    for (int s = 0; s < 2; s++) {  // two K=32 sub-chunks
      // A (16x32, M=bin, K=voxel): lanes<16 hold K {0-7,16-23}, lanes>=16 {8-15,24-31}
      v8h a0 = rowF[(s << 2) + halfSel];
      v8h a1 = rowF[(s << 2) + 2 + halfSel];
      v16h A = __builtin_shufflevector(a0, a1, 0, 1, 2, 3, 4, 5, 6, 7,
                                               8, 9, 10, 11, 12, 13, 14, 15);
      // B (32x16, K=voxel, N=bin): lanes<16 hold K {0-7,16-23}, lanes>=16 {8-15,24-31}
      v8h b0 = rowW[(s << 2) + halfSel];
      v8h b1 = rowW[(s << 2) + 2 + halfSel];
      v16h B = __builtin_shufflevector(b0, b1, 0, 1, 2, 3, 4, 5, 6, 7,
                                               8, 9, 10, 11, 12, 13, 14, 15);
      cj = __builtin_amdgcn_wmma_f32_16x16x32_f16(false, A,    false, B,    (short)0, cj, false, false);
      cf = __builtin_amdgcn_wmma_f32_16x16x32_f16(false, A,    false, ones, (short)0, cf, false, false);
      cw = __builtin_amdgcn_wmma_f32_16x16x32_f16(false, ones, false, B,    (short)0, cw, false, false);
    }
    LDS_ORDER_FENCE();  // WAR vs next iteration's stores (HW in-order; compiler fence)
  }

  // flush accumulators: C/D layout -> lane m = col, VGPR r -> row r + 8*halfSel
  float* hj = histj + b * 256;
#pragma unroll
  for (int r = 0; r < 8; r++)
    atomicAdd(&hj[(r + (halfSel << 3)) * 16 + m], cj[r]);
  if (m == 0) {
#pragma unroll
    for (int r = 0; r < 8; r++)
      atomicAdd(&histf[b * 16 + r + (halfSel << 3)], cf[r]);
  }
  if (halfSel == 0) atomicAdd(&histw[b * 16 + m], cw[0]);
}

// ---- kernel 3: normalize, entropies, MI, mean ----
__global__ void mi_final(const float* __restrict__ histj, const float* __restrict__ histf,
                         const float* __restrict__ histw, float* __restrict__ out) {
  __shared__ float mi[BATCH];
  const int t = threadIdx.x;
  if (t < BATCH) {
    const float* hj = histj + t * 256;
    float s = 0.0f;
    for (int i = 0; i < 256; i++) s += hj[i];
    float inv = 1.0f / (s + EPS_NORM);
    float hJ = 0.0f;
    for (int i = 0; i < 256; i++) { float p = hj[i] * inv + EPS_LOG; hJ -= p * __logf(p); }
    const float* hf = histf + t * 16;
    s = 0.0f; for (int i = 0; i < 16; i++) s += hf[i];
    inv = 1.0f / (s + EPS_NORM);
    float hF = 0.0f;
    for (int i = 0; i < 16; i++) { float p = hf[i] * inv + EPS_LOG; hF -= p * __logf(p); }
    const float* hw = histw + t * 16;
    s = 0.0f; for (int i = 0; i < 16; i++) s += hw[i];
    inv = 1.0f / (s + EPS_NORM);
    float hW = 0.0f;
    for (int i = 0; i < 16; i++) { float p = hw[i] * inv + EPS_LOG; hW -= p * __logf(p); }
    mi[t] = hF + hW - hJ;
  }
  __syncthreads();
  if (t == 0) out[0] = -0.25f * (mi[0] + mi[1] + mi[2] + mi[3]);
}

extern "C" void kernel_launch(void* const* d_in, const int* in_sizes, int n_in,
                              void* d_out, int out_size, void* d_ws, size_t ws_size,
                              hipStream_t stream) {
  const float* fixed  = (const float*)d_in[0];
  const float* warped = (const float*)d_in[1];
  const int n = in_sizes[0] / BATCH;  // voxels per batch (2,097,152)

  unsigned* mm = (unsigned*)d_ws;                       // [BATCH][4] minmax keys
  float* histj = (float*)((char*)d_ws + 64);            // [BATCH][256]
  float* histf = histj + BATCH * 256;                   // [BATCH][16]
  float* histw = histf + BATCH * 16;                    // [BATCH][16]
  float* out = (float*)d_out;

  mi_init<<<1, 256, 0, stream>>>(mm, histj, BATCH * (256 + 16 + 16));
  mi_minmax<<<dim3(64, BATCH), 256, 0, stream>>>(fixed, warped, mm, n);
  mi_hist<<<dim3(256, BATCH), 256, 0, stream>>>(fixed, warped, mm, histj, histf, histw, n);
  mi_final<<<1, 64, 0, stream>>>(histj, histf, histw, out);
}
